// TinyMambaHAR_17051020165228
// MI455X (gfx1250) — compile-verified, matrix operand order checked by reading, not run on hardware
//
#include <hip/hip_runtime.h>
#include <hip/hip_bf16.h>
#include <hip/hip_fp16.h>

// ---------------------------------------------------------------------------
// TinyMamba HAR forward for MI455X (gfx1250, wave32, WMMA)
//
// Pipeline:
//   pack x -> f16 (K padded 57->64); transpose+pad weights to f16
//   GEMM1: h   = x @ W_in + b_in            (WMMA f16, M=65536 N=64  K=64)
//   GEMM2: xr  = h @ in_proj_w              (WMMA f16, M=65536 N=256 K=64)
//   conv+silu: u = silu(causal dwconv(xi))  (VALU)
//   GEMM3: dbc = u @ x_proj_w               (WMMA f16, M=65536 N=48  K=128)
//   aux: delta = softplus(dt@dt_proj + b); sres = silu(res)
//   scan: per-(b,d,n) thread recurrence, reduce n-lanes via shfl_xor
//   head: out = (ysum @ out_proj_w) @ W_cls + b_cls   (mean folded into ysum)
// ---------------------------------------------------------------------------

typedef __attribute__((ext_vector_type(16))) _Float16 v16h;
typedef __attribute__((ext_vector_type(8)))  _Float16 v8h;
typedef __attribute__((ext_vector_type(8)))  float    v8f;

#define BATCH   64
#define LSEQ    1024
#define IN_DIM  57
#define DMODEL  64
#define DINNER  128
#define NSTATE  16
#define DTRANK  4
#define OUTCLS  6
#define BL      (BATCH * LSEQ)   // 65536 rows

// ---------------------------------------------------------------------------
// Pack x (b,l,57) f32 -> (BL,64) f16, zero-padded K.
__global__ void pack_x_f16(const float* __restrict__ x, _Float16* __restrict__ x16) {
    size_t idx = (size_t)blockIdx.x * blockDim.x + threadIdx.x;
    if (idx >= (size_t)BL * 64) return;
    int    k = (int)(idx & 63);
    size_t r = idx >> 6;
    x16[idx] = (k < IN_DIM) ? (_Float16)x[r * IN_DIM + k] : (_Float16)0.0f;
}

// Transpose (K,N) f32 weight -> (Npad,Kpad) f16, zero-padded.
__global__ void transpose_pad_f16(const float* __restrict__ W, int K, int N,
                                  int Kpad, int Npad, _Float16* __restrict__ WT) {
    size_t idx = (size_t)blockIdx.x * blockDim.x + threadIdx.x;
    if (idx >= (size_t)Npad * Kpad) return;
    int k = (int)(idx % Kpad);
    int n = (int)(idx / Kpad);
    _Float16 v = (_Float16)0.0f;
    if (k < K && n < N) v = (_Float16)W[(size_t)k * N + n];
    WT[idx] = v;
}

// ---------------------------------------------------------------------------
// WMMA GEMM: C[M,N] = A[M,K] * B[K,N] (+bias).  A row-major f16 (lda mult of 8),
// B given transposed (N-major, ldb = Kpad).  One 16x16 output tile per wave.
// Fragment layout matches ISA 16-bit A 16x32: lane L holds row M=L&15,
// halves K = (L>>4)*8 + {0..7} and +16 -> two contiguous b128 loads.
__global__ void wmma_gemm_f16(const _Float16* __restrict__ A, int lda,
                              const _Float16* __restrict__ BT, int ldb,
                              float* __restrict__ Cf, _Float16* __restrict__ Ch, int ldc,
                              const float* __restrict__ bias,
                              int ksteps, int ntiles) {
    const int lane  = threadIdx.x & 31;
    const int wave  = threadIdx.x >> 5;
    const int ntile = blockIdx.y * (blockDim.x >> 5) + wave;
    if (ntile >= ntiles) return;
    const int mtile = blockIdx.x;
    const int khalf = (lane >> 4) << 3;             // 0 or 8

    const _Float16* Ap = A  + (size_t)(mtile * 16 + (lane & 15)) * lda + khalf;
    const _Float16* Bp = BT + (size_t)(ntile * 16 + (lane & 15)) * ldb + khalf;

    v8f acc{};
    for (int ks = 0; ks < ksteps; ++ks) {
        const _Float16* ap = Ap + ks * 32;
        v8h alo = *(const v8h*)(ap);
        v8h ahi = *(const v8h*)(ap + 16);
        const _Float16* bp = Bp + ks * 32;
        v8h blo = *(const v8h*)(bp);
        v8h bhi = *(const v8h*)(bp + 16);
        v16h a = __builtin_shufflevector(alo, ahi, 0,1,2,3,4,5,6,7,8,9,10,11,12,13,14,15);
        v16h b = __builtin_shufflevector(blo, bhi, 0,1,2,3,4,5,6,7,8,9,10,11,12,13,14,15);
        acc = __builtin_amdgcn_wmma_f32_16x16x32_f16(
            /*neg_a=*/false, a, /*neg_b=*/false, b,
            /*c_mod=*/(short)0, acc, /*reuse_a=*/false, /*reuse_b=*/false);
    }

    const int   col = ntile * 16 + (lane & 15);
    const int   r0  = mtile * 16 + ((lane >> 4) << 3);
    const float bv  = bias ? bias[col] : 0.0f;
#pragma unroll
    for (int i = 0; i < 8; ++i) {
        float v = acc[i] + bv;
        if (Cf) Cf[(size_t)(r0 + i) * ldc + col] = v;
        if (Ch) Ch[(size_t)(r0 + i) * ldc + col] = (_Float16)v;
    }
}

// ---------------------------------------------------------------------------
// Causal depthwise conv (D_CONV=4) over time + SiLU.  xi = xr[..., 0:128].
__global__ void conv_silu_kernel(const float* __restrict__ xr,
                                 const float* __restrict__ cw,   // (128,1,4)
                                 const float* __restrict__ cb,   // (128)
                                 float* __restrict__ u32, _Float16* __restrict__ u16) {
    size_t idx = (size_t)blockIdx.x * blockDim.x + threadIdx.x;
    if (idx >= (size_t)BL * DINNER) return;
    int    d  = (int)(idx & (DINNER - 1));
    size_t bl = idx >> 7;
    int    l  = (int)(bl & (LSEQ - 1));
    const float* w = cw + d * 4;
    float s = cb[d];
#pragma unroll
    for (int j = 0; j < 4; ++j) {
        int lj = l - 3 + j;
        if (lj >= 0)
            s = fmaf(xr[(bl - (size_t)(3 - j)) * 256 + d], w[j], s);
    }
    float v = s / (1.0f + __expf(-s));   // silu
    u32[idx] = v;
    u16[idx] = (_Float16)v;
}

// ---------------------------------------------------------------------------
// delta = softplus(dt @ dt_proj_w + dt_proj_b); sres = silu(res).
__global__ void aux_delta_sres(const float* __restrict__ dbc,   // (BL,48)
                               const float* __restrict__ xr,    // (BL,256)
                               const float* __restrict__ dtw,   // (4,128)
                               const float* __restrict__ dtb,   // (128)
                               float* __restrict__ delta, float* __restrict__ sres) {
    size_t idx = (size_t)blockIdx.x * blockDim.x + threadIdx.x;
    if (idx >= (size_t)BL * DINNER) return;
    int    d  = (int)(idx & (DINNER - 1));
    size_t bl = idx >> 7;
    const float* row = dbc + bl * 48;
    float dt = dtb[d];
    dt = fmaf(row[0], dtw[0 * DINNER + d], dt);
    dt = fmaf(row[1], dtw[1 * DINNER + d], dt);
    dt = fmaf(row[2], dtw[2 * DINNER + d], dt);
    dt = fmaf(row[3], dtw[3 * DINNER + d], dt);
    delta[idx] = (dt > 20.0f) ? dt : __logf(1.0f + __expf(dt));
    float r = xr[bl * 256 + DINNER + d];
    sres[idx] = r / (1.0f + __expf(-r));
}

// ---------------------------------------------------------------------------
// Selective scan, one thread per (b, d, n).  h_n recurrence is independent
// per n; silu(res)-weighted output is accumulated per thread and the n-sum
// is done once at the end with xor-shuffles inside each 16-lane half.
__global__ void scan_kernel(const float* __restrict__ u,       // (b,l,128)
                            const float* __restrict__ delta,   // (b,l,128)
                            const float* __restrict__ sres,    // (b,l,128)
                            const float* __restrict__ dbc,     // (b,l,48): [4..19]=B, [20..35]=C
                            const float* __restrict__ A_log,   // (128,16)
                            const float* __restrict__ Dvec,    // (128)
                            float* __restrict__ ysum) {        // (b,128), includes 1/L
    const int n    = threadIdx.x & 15;
    const int dloc = threadIdx.x >> 4;                 // 0..15
    const int b    = blockIdx.x >> 3;
    const int d    = ((blockIdx.x & 7) << 4) + dloc;

    const float An = -__expf(A_log[d * NSTATE + n]);
    const float Dd = (n == 0) ? Dvec[d] : 0.0f;        // add u*D once per d

    const float* ub = u     + (size_t)b * LSEQ * DINNER + d;
    const float* db = delta + (size_t)b * LSEQ * DINNER + d;
    const float* sb = sres  + (size_t)b * LSEQ * DINNER + d;
    const float* cb = dbc   + (size_t)b * LSEQ * 48;

    float h = 0.0f, p = 0.0f;
    for (int l = 0; l < LSEQ; ++l) {
        const float* row = cb + l * 48;
        __builtin_prefetch(row + 48, 0, 1);            // global_prefetch_b8
        float dt = db[(size_t)l * DINNER];
        float ut = ub[(size_t)l * DINNER];
        float sr = sb[(size_t)l * DINNER];
        float Bn = row[4 + n];
        float Cn = row[20 + n];
        float dA = __expf(dt * An);
        h = __builtin_fmaf(dA, h, dt * Bn * ut);
        float yc = __builtin_fmaf(h, Cn, ut * Dd);
        p = __builtin_fmaf(sr, yc, p);
    }
    // reduce across the 16 n-lanes (masks 1/2/4/8 stay within a 16-lane half)
    p += __shfl_xor(p, 1);
    p += __shfl_xor(p, 2);
    p += __shfl_xor(p, 4);
    p += __shfl_xor(p, 8);
    if (n == 0) ysum[b * DINNER + d] = p * (1.0f / (float)LSEQ);
}

// ---------------------------------------------------------------------------
// out = (ysum @ out_proj_w) @ W_cls + b_cls   (ysum already time-averaged)
__global__ void head_kernel(const float* __restrict__ ysum,   // (b,128)
                            const float* __restrict__ outw,   // (128,64)
                            const float* __restrict__ wcls,   // (64,6)
                            const float* __restrict__ bcls,   // (6)
                            float* __restrict__ out) {        // (b,6)
    __shared__ float pooled[DMODEL];
    const int b  = blockIdx.x;
    const int dm = threadIdx.x;                       // 64 threads
    float s = 0.0f;
    for (int di = 0; di < DINNER; ++di)
        s = fmaf(ysum[b * DINNER + di], outw[(size_t)di * DMODEL + dm], s);
    pooled[dm] = s;
    __syncthreads();
    if (dm < OUTCLS) {
        float o = bcls[dm];
        for (int k = 0; k < DMODEL; ++k)
            o = fmaf(pooled[k], wcls[k * OUTCLS + dm], o);
        out[b * OUTCLS + dm] = o;
    }
}

// ---------------------------------------------------------------------------
extern "C" void kernel_launch(void* const* d_in, const int* in_sizes, int n_in,
                              void* d_out, int out_size, void* d_ws, size_t ws_size,
                              hipStream_t stream) {
    (void)in_sizes; (void)n_in; (void)out_size; (void)ws_size;

    const float* x     = (const float*)d_in[0];
    const float* W_in  = (const float*)d_in[1];
    const float* b_in  = (const float*)d_in[2];
    const float* ipw   = (const float*)d_in[3];
    const float* convw = (const float*)d_in[4];
    const float* convb = (const float*)d_in[5];
    const float* xpw   = (const float*)d_in[6];
    const float* dtw   = (const float*)d_in[7];
    const float* dtb   = (const float*)d_in[8];
    const float* A_log = (const float*)d_in[9];
    const float* Dv    = (const float*)d_in[10];
    const float* outw  = (const float*)d_in[11];
    const float* wcls  = (const float*)d_in[12];
    const float* bcls  = (const float*)d_in[13];
    float* out = (float*)d_out;

    // workspace carve-out (256B aligned)
    char*  ws  = (char*)d_ws;
    size_t off = 0;
    auto take = [&](size_t bytes) -> char* {
        char* p = ws + off;
        off = (off + bytes + 255) & ~(size_t)255;
        return p;
    };
    _Float16* x16  = (_Float16*)take((size_t)BL * 64 * 2);     // x padded f16
    _Float16* h16  = (_Float16*)take((size_t)BL * 64 * 2);     // h = x@W_in+b
    _Float16* u16  = (_Float16*)take((size_t)BL * 128 * 2);    // u f16 (GEMM3 A)
    _Float16* winT = (_Float16*)take((size_t)64 * 64 * 2);     // W_in^T pad
    _Float16* ipT  = (_Float16*)take((size_t)256 * 64 * 2);    // in_proj^T
    _Float16* xpT  = (_Float16*)take((size_t)48 * 128 * 2);    // x_proj^T pad
    float* xr    = (float*)take((size_t)BL * 256 * 4);         // [xi | res]
    float* u32   = (float*)take((size_t)BL * 128 * 4);
    float* dbcb  = (float*)take((size_t)BL * 48  * 4);         // [dt4 | B16 | C16 | pad]
    float* delta = (float*)take((size_t)BL * 128 * 4);
    float* sres  = (float*)take((size_t)BL * 128 * 4);
    float* ysum  = (float*)take((size_t)BATCH * 128 * 4);

    // pack / transpose
    pack_x_f16<<<(BL * 64) / 256, 256, 0, stream>>>(x, x16);
    transpose_pad_f16<<<(64 * 64) / 256, 256, 0, stream>>>(W_in, IN_DIM, 64, 64, 64, winT);
    transpose_pad_f16<<<(256 * 64) / 256, 256, 0, stream>>>(ipw, 64, 256, 64, 256, ipT);
    transpose_pad_f16<<<(48 * 128) / 256, 256, 0, stream>>>(xpw, 128, 36, 128, 48, xpT);

    // GEMM1: h16 = x @ W_in + b_in   (M=65536, Ntiles=4, K=64)
    wmma_gemm_f16<<<dim3(BL / 16, 1), 128, 0, stream>>>(
        x16, 64, winT, 64, nullptr, h16, 64, b_in, /*ksteps=*/2, /*ntiles=*/4);
    // GEMM2: xr = h @ in_proj_w      (Ntiles=16, K=64)
    wmma_gemm_f16<<<dim3(BL / 16, 2), 256, 0, stream>>>(
        h16, 64, ipT, 64, xr, nullptr, 256, nullptr, /*ksteps=*/2, /*ntiles=*/16);
    // conv + silu
    conv_silu_kernel<<<(BL * 128) / 256, 256, 0, stream>>>(xr, convw, convb, u32, u16);
    // GEMM3: dbc = u @ x_proj_w      (Ntiles=3, K=128)
    wmma_gemm_f16<<<dim3(BL / 16, 1), 96, 0, stream>>>(
        u16, 128, xpT, 128, dbcb, nullptr, 48, nullptr, /*ksteps=*/4, /*ntiles=*/3);
    // delta + silu(res)
    aux_delta_sres<<<(BL * 128) / 256, 256, 0, stream>>>(dbcb, xr, dtw, dtb, delta, sres);
    // selective scan (64*8 blocks x 256 threads = 131072 threads / 4096 waves)
    scan_kernel<<<BATCH * 8, 256, 0, stream>>>(u32, delta, sres, dbcb, A_log, Dv, ysum);
    // head
    head_kernel<<<BATCH, 64, 0, stream>>>(ysum, outw, wcls, bcls, out);
}
